// AutoCorrelation_25512105738563
// MI455X (gfx1250) — matrix operand order repaired and checked by reading, not compiled, not run on hardware
//
#include <hip/hip_runtime.h>
#include <hip/hip_bf16.h>

// ---------------------------------------------------------------------------
// AutoCorrelation (Autoformer) for MI455X / gfx1250.
// B=32, L=1536, H=8, C=64 (HC=512), TOPK = int(3*ln(1536)) = 22.
// Stage 1: mean_value[b,tau] = (1/512) * sum_j  q_row[b,(j+tau)%L] . k_row[b,j]
//          == lag-diagonal reduction of batched GEMM Q(LxHC) x K^T(HCxL).
//          v_wmma_f32_16x16x32_f16 tiles; K tiles streamed by the Tensor Data
//          Mover (tensor_load_to_lds, double buffered, s_wait_tensorcnt).
// Stage 2: batch mean, top-22 lags, per-batch softmax.
// Stage 3: out[b,t] = sum_d w[b,d] * value[b,(t+idx[d])%L]  (float4 stream).
// ---------------------------------------------------------------------------

#define AC_B    32
#define AC_L    1536
#define AC_HC   512
#define AC_TOPK 22
#define AC_ROWP (AC_HC + 8)          // padded fp16 LDS row: conflict-free b128 reads
#define AC_BAND 64                   // i-rows per workgroup (4 waves x 16)
#define AC_NJ   (AC_L / 16)          // 96 K tiles
#define AC_KTILE_ELTS (16 * AC_HC)   // 8192 f32 per K tile (contiguous in global)

typedef __attribute__((ext_vector_type(16))) _Float16 v16h;
typedef __attribute__((ext_vector_type(8)))  _Float16 v8h;
typedef __attribute__((ext_vector_type(4)))  _Float16 v4h;
typedef __attribute__((ext_vector_type(8)))  float    v8f;
typedef __attribute__((ext_vector_type(4)))  unsigned int v4u;
typedef __attribute__((ext_vector_type(8)))  int      v8i;
typedef __attribute__((ext_vector_type(4)))  int      v4i;

// ---------------------------------------------------------------------------
// TDM: DMA one contiguous 16-row fp32 K tile (32 KB) from global into LDS.
// D# group0: count=1 | lds_addr | global_addr(57b) | type=2
// D# group1: data_size=4B, tensor_dim0 = tile_dim0 = stride = 8192, 1 line.
// ---------------------------------------------------------------------------
__device__ __forceinline__ void ac_tdm_load_ktile(const float* gsrc,
                                                  unsigned int lds_off) {
    const unsigned long long ga = (unsigned long long)(size_t)gsrc;
    v4u g0;
    g0[0] = 1u;                                          // count=1 (valid), user
    g0[1] = lds_off;                                     // LDS byte address
    g0[2] = (unsigned int)(ga & 0xFFFFFFFFu);            // global_addr[31:0]
    g0[3] = ((unsigned int)(ga >> 32) & 0x01FFFFFFu)     // global_addr[56:32]
          | 0x80000000u;                                 // type=2 ("image")
    v8i g1;
    g1[0] = 0x00020000;                  // wg_mask=0 | data_size=2 (4B) | flags=0
    g1[1] = (int)(8192u << 16);          // tensor_dim0[15:0]=8192 (atomic_bar=0)
    g1[2] = 0x00010000;                  // tensor_dim0[31:16]=0 | tensor_dim1=1
    g1[3] = (int)(8192u << 16);          // tensor_dim1 hi=0 | tile_dim0=8192
    g1[4] = 1;                           // tile_dim1=1 | tile_dim2=0
    g1[5] = 8192;                        // tensor_dim0_stride[31:0]
    g1[6] = 0;                           // stride hi | tensor_dim1_stride lo
    g1[7] = 0;                           // tensor_dim1_stride hi
    const v4i gz = {0, 0, 0, 0};
#if __clang_major__ >= 23
    const v8i gz8 = {0, 0, 0, 0, 0, 0, 0, 0};
    __builtin_amdgcn_tensor_load_to_lds(g0, g1, gz, gz, gz8, 0);
#else
    __builtin_amdgcn_tensor_load_to_lds(g0, g1, gz, gz, 0);
#endif
}

__device__ __forceinline__ void ac_wait_tensorcnt0() {
#if __has_builtin(__builtin_amdgcn_s_wait_tensorcnt)
    __builtin_amdgcn_s_wait_tensorcnt((short)0);
#else
    asm volatile("s_wait_tensorcnt 0" ::: "memory");
#endif
}

// ---------------------------------------------------------------------------
// Stage 1: correlation GEMM with lag-diagonal scatter.
// grid = (L/AC_BAND, B), block = 128 (4 waves).  Dynamic LDS:
//   qtile : 64 x AC_ROWP halfs   (Q band, fp16)            66560 B
//   ktile : 16 x AC_ROWP halfs   (current K tile, fp16)    16640 B
//   kstage: 2 x 8192 f32         (TDM double buffer)       65536 B
//   lag   : AC_L floats          (lag accumulator)          6144 B
// ---------------------------------------------------------------------------
__launch_bounds__(128)
__global__ void ac_corr_wmma_kernel(const float* __restrict__ q,
                                    const float* __restrict__ k,
                                    float* __restrict__ mean_value) {
    extern __shared__ char smem_raw[];
    _Float16* qtile  = reinterpret_cast<_Float16*>(smem_raw);
    _Float16* ktile  = qtile + AC_BAND * AC_ROWP;
    float*    kstage = reinterpret_cast<float*>(ktile + 16 * AC_ROWP);
    float*    lag    = kstage + 2 * AC_KTILE_ELTS;

    const int tid  = threadIdx.x;
    const int lane = tid & 31;
    const int wave = tid >> 5;             // 0..3
    const int b    = blockIdx.y;
    const int I0   = blockIdx.x * AC_BAND; // first i-row of this band

    // Kick off TDM for K tile J=0 (wave 0 only; TDM ignores EXEC, cnt per-wave)
    const float* kbase = k + (size_t)b * AC_L * AC_HC;
    if (wave == 0) {
        ac_tdm_load_ktile(kbase, (unsigned int)(size_t)kstage);
    }

    // zero the lag accumulator
    for (int t = tid; t < AC_L; t += 128) lag[t] = 0.0f;

    // Load Q band: 64 rows x 512 f32 -> fp16 LDS (float4 loads, packed stores)
    const float4* q4 = reinterpret_cast<const float4*>(q);
    for (int e = tid; e < AC_BAND * (AC_HC / 4); e += 128) {
        const int row  = e >> 7;           // /128
        const int col4 = e & 127;
        float4 v = q4[(size_t)(b * AC_L + I0 + row) * (AC_HC / 4) + col4];
        v4h h = { (_Float16)v.x, (_Float16)v.y, (_Float16)v.z, (_Float16)v.w };
        *reinterpret_cast<v4h*>(qtile + row * AC_ROWP + col4 * 4) = h;
    }

    const int rowM = lane & 15;            // A-frag row / B-frag column
    const int hi   = lane >> 4;            // lane half selector

    for (int J = 0; J < AC_NJ; ++J) {
        __syncthreads();                   // compute J-1 done; stage[(J+1)&1] free
        if (wave == 0) {
            ac_wait_tensorcnt0();          // stage[J&1] landed
            if (J + 1 < AC_NJ) {           // overlap next DMA with cvt + compute
                ac_tdm_load_ktile(kbase + (size_t)(J + 1) * AC_KTILE_ELTS,
                                  (unsigned int)(size_t)(kstage
                                      + ((J + 1) & 1) * AC_KTILE_ELTS));
            }
        }
        __syncthreads();                   // publish stage[J&1] to all waves

        // Convert staged fp32 tile -> padded fp16 ktile (LDS -> LDS)
        const float4* s4 = reinterpret_cast<const float4*>(
            kstage + (J & 1) * AC_KTILE_ELTS);
        for (int e = tid; e < 16 * (AC_HC / 4); e += 128) {
            const int row  = e >> 7;
            const int col4 = e & 127;
            float4 v = s4[row * (AC_HC / 4) + col4];
            v4h h = { (_Float16)v.x, (_Float16)v.y, (_Float16)v.z, (_Float16)v.w };
            *reinterpret_cast<v4h*>(ktile + row * AC_ROWP + col4 * 4) = h;
        }
        __syncthreads();

        // 16x16 output tile (i = I0 + wave*16 + M, j = J*16 + N), K-depth 512
        v8f acc = {};
        const _Float16* qrow = qtile + (wave * 16 + rowM) * AC_ROWP;
        const _Float16* krow = ktile + rowM * AC_ROWP;
#pragma unroll
        for (int kk = 0; kk < AC_HC / 32; ++kk) {
            const int kb = kk * 32;
            // A fragment (16x32 f16): lane half picks K {0..7,16..23}/{8..15,24..31}
            v8h a0 = *reinterpret_cast<const v8h*>(qrow + kb + hi * 8);
            v8h a1 = *reinterpret_cast<const v8h*>(qrow + kb + 16 + hi * 8);
            v16h av = __builtin_shufflevector(a0, a1, 0, 1, 2, 3, 4, 5, 6, 7,
                                              8, 9, 10, 11, 12, 13, 14, 15);
            // B fragment (32x16 f16) = K-tile row as a column: K {0..7,16..23}/{8..15,24..31}
            v8h b0 = *reinterpret_cast<const v8h*>(krow + kb + hi * 8);
            v8h b1 = *reinterpret_cast<const v8h*>(krow + kb + 16 + hi * 8);
            v16h bv = __builtin_shufflevector(b0, b1, 0, 1, 2, 3, 4, 5, 6, 7,
                                              8, 9, 10, 11, 12, 13, 14, 15);
            acc = __builtin_amdgcn_wmma_f32_16x16x32_f16(
                false, av, false, bv, (short)0, acc, false, false);
        }

        // Scatter the 16x16 tile onto lag diagonals: tau = (i - j) mod L
        const int jcol = J * 16 + rowM;    // N = lane & 15
#pragma unroll
        for (int r = 0; r < 8; ++r) {
            const int mrow = r + (hi << 3);            // C-layout: lanes>=16 -> M+8
            const int irow = I0 + wave * 16 + mrow;
            int tau = irow - jcol;
            tau = (tau < 0) ? tau + AC_L : tau;
            atomicAdd(&lag[tau], acc[r]);              // ds_add_f32
        }
    }

    __syncthreads();
    for (int t = tid; t < AC_L; t += 128)
        atomicAdd(&mean_value[b * AC_L + t], lag[t] * (1.0f / (float)AC_HC));
}

// ---------------------------------------------------------------------------
// Stage 2a: meanL[tau] = mean over batch of mean_value[:, tau]
// ---------------------------------------------------------------------------
__global__ void ac_meanl_kernel(const float* __restrict__ mean_value,
                                float* __restrict__ meanL) {
    const int t = blockIdx.x * blockDim.x + threadIdx.x;
    if (t >= AC_L) return;
    float s = 0.0f;
    for (int b = 0; b < AC_B; ++b) s += mean_value[b * AC_L + t];
    meanL[t] = s * (1.0f / (float)AC_B);
}

// ---------------------------------------------------------------------------
// Stage 2b: top-22 lags via iterative argmax (single block).
// ---------------------------------------------------------------------------
__launch_bounds__(256)
__global__ void ac_topk_kernel(const float* __restrict__ meanL,
                               int* __restrict__ idx_topk) {
    __shared__ float sv[AC_L];
    __shared__ float rb[256];
    __shared__ int   ri[256];
    const int tid = threadIdx.x;
    for (int t = tid; t < AC_L; t += 256) sv[t] = meanL[t];
    __syncthreads();
    for (int sel = 0; sel < AC_TOPK; ++sel) {
        float best = -3.4e38f; int bi = 0;
        for (int t = tid; t < AC_L; t += 256)
            if (sv[t] > best) { best = sv[t]; bi = t; }
        rb[tid] = best; ri[tid] = bi;
        __syncthreads();
        for (int s = 128; s > 0; s >>= 1) {
            if (tid < s && rb[tid + s] > rb[tid]) {
                rb[tid] = rb[tid + s]; ri[tid] = ri[tid + s];
            }
            __syncthreads();
        }
        if (tid == 0) { idx_topk[sel] = ri[0]; sv[ri[0]] = -3.4e38f; }
        __syncthreads();
    }
}

// ---------------------------------------------------------------------------
// Stage 2c: per-batch softmax over the selected lag scores.
// ---------------------------------------------------------------------------
__global__ void ac_softmax_kernel(const float* __restrict__ mean_value,
                                  const int* __restrict__ idx_topk,
                                  float* __restrict__ wts) {
    const int b = threadIdx.x;
    if (b >= AC_B) return;
    float m[AC_TOPK];
    float mx = -3.4e38f;
    for (int d = 0; d < AC_TOPK; ++d) {
        m[d] = mean_value[b * AC_L + idx_topk[d]];
        mx = fmaxf(mx, m[d]);
    }
    float s = 0.0f;
    for (int d = 0; d < AC_TOPK; ++d) { m[d] = __expf(m[d] - mx); s += m[d]; }
    const float inv = 1.0f / s;
    for (int d = 0; d < AC_TOPK; ++d) wts[b * 32 + d] = m[d] * inv;
}

// ---------------------------------------------------------------------------
// Stage 3: out[b,t,:] = sum_d w[b,d] * value[b,(t+idx[d])%L,:]   (float4)
// grid over B*L*(HC/4); one block never crosses a batch boundary.
// ---------------------------------------------------------------------------
__launch_bounds__(256)
__global__ void ac_gather_kernel(const float* __restrict__ value,
                                 const int* __restrict__ idx_topk,
                                 const float* __restrict__ wts,
                                 float* __restrict__ out) {
    const int gid  = blockIdx.x * 256 + threadIdx.x;  // over B*L*128
    const int b    = gid / (AC_L * 128);
    const int rem  = gid - b * (AC_L * 128);
    const int t    = rem >> 7;
    const int hc4  = rem & 127;

    __shared__ int   sidx[AC_TOPK];
    __shared__ float sw[AC_TOPK];
    if (threadIdx.x < AC_TOPK) {
        sidx[threadIdx.x] = idx_topk[threadIdx.x];
        sw[threadIdx.x]   = wts[b * 32 + threadIdx.x];
    }
    __syncthreads();

    const float4* v4 = reinterpret_cast<const float4*>(value);
    float4 acc; acc.x = acc.y = acc.z = acc.w = 0.0f;
#pragma unroll 1
    for (int d = 0; d < AC_TOPK; ++d) {
        int td = t + sidx[d];
        if (td >= AC_L) td -= AC_L;
        if (d + 1 < AC_TOPK) {                        // warm L2 for next lag row
            int tn = t + sidx[d + 1];
            if (tn >= AC_L) tn -= AC_L;
            __builtin_prefetch(&v4[((size_t)b * AC_L + tn) * 128 + hc4], 0, 1);
        }
        float4 vv = v4[((size_t)b * AC_L + td) * 128 + hc4];
        const float w = sw[d];
        acc.x += w * vv.x; acc.y += w * vv.y;
        acc.z += w * vv.z; acc.w += w * vv.w;
    }
    reinterpret_cast<float4*>(out)[gid] = acc;
}

// ---------------------------------------------------------------------------
extern "C" void kernel_launch(void* const* d_in, const int* in_sizes, int n_in,
                              void* d_out, int out_size, void* d_ws, size_t ws_size,
                              hipStream_t stream) {
    (void)in_sizes; (void)n_in; (void)out_size; (void)ws_size;
    const float* q = (const float*)d_in[0];
    const float* k = (const float*)d_in[1];
    const float* v = (const float*)d_in[2];
    float* out = (float*)d_out;

    // Workspace layout (floats): mean_value[B*L] | meanL[L] | idx[32] | wts[B*32]
    float* mean_value = (float*)d_ws;
    float* meanL      = mean_value + AC_B * AC_L;
    int*   idx_topk   = (int*)(meanL + AC_L);
    float* wts        = (float*)(idx_topk + 32);

    hipMemsetAsync(mean_value, 0, (size_t)AC_B * AC_L * sizeof(float), stream);

    const size_t smem = (size_t)(AC_BAND + 16) * AC_ROWP * sizeof(_Float16)
                      + (size_t)2 * AC_KTILE_ELTS * sizeof(float)
                      + (size_t)AC_L * sizeof(float);
    dim3 grid1(AC_L / AC_BAND, AC_B);
    ac_corr_wmma_kernel<<<grid1, 128, smem, stream>>>(q, k, mean_value);

    ac_meanl_kernel<<<(AC_L + 255) / 256, 256, 0, stream>>>(mean_value, meanL);
    ac_topk_kernel<<<1, 256, 0, stream>>>(meanL, idx_topk);
    ac_softmax_kernel<<<1, 64, 0, stream>>>(mean_value, idx_topk, wts);

    const int total4 = AC_B * AC_L * (AC_HC / 4);     // 6,291,456
    ac_gather_kernel<<<total4 / 256, 256, 0, stream>>>(v, idx_topk, wts, out);
}
